// PatchNCELoss_73177652789404
// MI455X (gfx1250) — matrix operand order repaired and checked by reading, not compiled, not run on hardware
//
#include <hip/hip_runtime.h>
#include <hip/hip_bf16.h>

// ---------------------------------------------------------------------------
// PatchNCELoss  (B=8, C=256, HW=1024)  ->  scalar fp32 loss
// Strategy: bf16 WMMA (v_wmma_f32_16x16x32_bf16) for both GEMMs,
// fused logsumexp epilogue on the big (8192x8192x256) l_neg GEMM.
// ---------------------------------------------------------------------------

typedef __attribute__((ext_vector_type(16))) __bf16 v16bf;
typedef __attribute__((ext_vector_type(8)))  __bf16 v8bf;
typedef __attribute__((ext_vector_type(8)))  float  v8f;

#define B_    8
#define C_    256
#define HW_   1024
#define M_    8192           // B*HW
#define CHW_  (C_*HW_)       // 262144
#define INV_T 14.285714285714286f
#define NEGF  (-10.0f)

// ---------------------------------------------------------------------------
// Kernel A: per-column inverse L2 norms over channel dim.
// grid = 32 blocks x 256 thr; block -> (b, 256 consecutive hw). Coalesced.
// ---------------------------------------------------------------------------
__global__ __launch_bounds__(256) void norms_kernel(
    const float* __restrict__ fq, const float* __restrict__ fk,
    float* __restrict__ invq, float* __restrict__ invk)
{
    const int b = blockIdx.x >> 2;
    const int h = ((blockIdx.x & 3) << 8) + threadIdx.x;
    const float* q = fq + b * CHW_ + h;
    const float* k = fk + b * CHW_ + h;
    float sq = 0.f, sk = 0.f;
    #pragma unroll 4
    for (int c = 0; c < C_; ++c) {
        float a = q[c * HW_]; sq += a * a;
        float d = k[c * HW_]; sk += d * d;
    }
    invq[b * HW_ + h] = __frsqrt_rn(sq);
    invk[b * HW_ + h] = __frsqrt_rn(sk);
}

// ---------------------------------------------------------------------------
// Kernel B: build the four bf16 operand buffers, each [8192 x 256] row-major
// so WMMA fragments become contiguous per-lane loads.
//   Ap  [b*1024+h][c] = qn[b,c,h]                      (l_pos A)
//   BpT [b*1024+k][c] = kn[b,c,k]                      (l_pos B, col-major)
//   Al  [m][c]        = fq_t[m,c]   (scrambled gather) (l_neg A)
//   BlT [n][c]        = fk_flat[c,n]                   (l_neg B, col-major)
// grid = 8192 blocks x 256 thr ; i = row*256 + c.
// ---------------------------------------------------------------------------
__global__ __launch_bounds__(256) void prep_kernel(
    const float* __restrict__ fq, const float* __restrict__ fk,
    const float* __restrict__ invq, const float* __restrict__ invk,
    __bf16* __restrict__ Ap, __bf16* __restrict__ BpT,
    __bf16* __restrict__ Al, __bf16* __restrict__ BlT)
{
    const int i = blockIdx.x * 256 + threadIdx.x;   // [0, 2M)
    const int m = i >> 8;                           // row   [0, 8192)
    const int c = i & 255;                          // col   [0, 256)
    {   // Ap
        const int b = m >> 10, h = m & 1023;
        Ap[i] = (__bf16)(fq[b * CHW_ + c * HW_ + h] * invq[m]);
    }
    {   // BpT
        const int b = m >> 10, k = m & 1023;
        BpT[i] = (__bf16)(fk[b * CHW_ + c * HW_ + k] * invk[m]);
    }
    {   // Al : fq_t[m,c] = qn[c&7, (m&7)*32 + (c>>3), m>>3]
        const int bb = c & 7, cc = ((m & 7) << 5) + (c >> 3), hh = m >> 3;
        Al[i] = (__bf16)(fq[bb * CHW_ + cc * HW_ + hh] * invq[bb * HW_ + hh]);
    }
    {   // BlT : fk_flat[c,n] = kn[c>>5, (c&31)*8 + (n>>10), n&1023],  n = m
        const int n = m;
        const int bb = c >> 5, cc = ((c & 31) << 3) + (n >> 10), hh = n & 1023;
        BlT[i] = (__bf16)(fk[bb * CHW_ + cc * HW_ + hh] * invk[bb * HW_ + hh]);
    }
}

// ---------------------------------------------------------------------------
// WMMA helpers.  A fragment (16x32 bf16): lane = (half = L>>4, mrow = L&15);
//   elems 0..15 -> K = kk*32 + 16*half + e
// B fragment (32x16 bf16): lane = (half, ncol = L&15);
//   elems 0..15 -> K = kk*32 + 16*half + e   (one contiguous 32B load)
// ---------------------------------------------------------------------------
__device__ __forceinline__ void load_a_frags(const __bf16* __restrict__ arow,
                                             int half, v16bf (&af)[8])
{
    #pragma unroll
    for (int kk = 0; kk < 8; ++kk) {
        v8bf lo = *(const v8bf*)(arow + kk * 32 + 16 * half);
        v8bf hi = *(const v8bf*)(arow + kk * 32 + 16 * half + 8);
        af[kk] = __builtin_shufflevector(lo, hi,
                    0,1,2,3,4,5,6,7,8,9,10,11,12,13,14,15);
    }
}

__device__ __forceinline__ v8f wmma_k256(const v16bf (&af)[8],
                                         const __bf16* __restrict__ brow,
                                         int half)
{
    v8f acc = {0.f,0.f,0.f,0.f,0.f,0.f,0.f,0.f};
    #pragma unroll
    for (int kk = 0; kk < 8; ++kk) {
        v16bf bf = *(const v16bf*)(brow + kk * 32 + 16 * half);
        acc = __builtin_amdgcn_wmma_f32_16x16x32_bf16(
                  false, af[kk], false, bf, (short)0, acc, false, false);
    }
    return acc;
}

// ---------------------------------------------------------------------------
// Kernel C: l_pos GEMM.  P[b*1024+h][k] = sum_c qn[b,c,h]*kn[b,c,k]
// grid = 64 blocks x 256 thr (8 waves). Each wave: 16 h-rows, loops 64
// k-strips of 16, K=256 as 8 chained WMMAs.
// ---------------------------------------------------------------------------
__global__ __launch_bounds__(256) void pos_gemm_kernel(
    const __bf16* __restrict__ Ap, const __bf16* __restrict__ BpT,
    float* __restrict__ P)
{
    const int lane  = threadIdx.x & 31;
    const int wave  = threadIdx.x >> 5;
    const int gwave = blockIdx.x * 8 + wave;      // [0, 512)
    const int b     = gwave >> 6;                 // [0, 8)
    const int h0    = (gwave & 63) << 4;          // [0, 1024) step 16
    const int half  = lane >> 4;
    const int mrow  = lane & 15;
    const int ncol  = lane & 15;

    v16bf af[8];
    load_a_frags(Ap + (size_t)(b * HW_ + h0 + mrow) * C_, half, af);

    for (int k0 = 0; k0 < HW_; k0 += 16) {
        const __bf16* brow = BpT + (size_t)(b * HW_ + k0 + ncol) * C_;
        v8f acc = wmma_k256(af, brow, half);
        #pragma unroll
        for (int v = 0; v < 8; ++v) {
            const int row = b * HW_ + h0 + v + 8 * half;
            P[(size_t)row * HW_ + k0 + ncol] = acc[v];
        }
    }
}

// ---------------------------------------------------------------------------
// Kernel D: l_neg GEMM fused with logsumexp epilogue.
// L[m][n] over 8192x8192, K=256.  Row r=(b,h,k) logits are
// [P[m][k], L[m][8k..8k+7]]/T with diag(m==n) -> -10.  Normalized inputs
// bound |logit| <= ~14.3, so plain exp/log is numerically safe.
// Column strips of 16 cover exactly 2 k-groups; each k-group = one
// xor-closed 8-lane octet of the WMMA C layout -> shfl_xor(1,2,4) reduce.
// grid = 64 blocks x 256 thr; wave owns 16 m-rows, loops all 512 strips.
// ---------------------------------------------------------------------------
__global__ __launch_bounds__(256) void lneg_lse_kernel(
    const __bf16* __restrict__ Al, const __bf16* __restrict__ BlT,
    const float* __restrict__ P, double* __restrict__ partials)
{
    const int lane  = threadIdx.x & 31;
    const int wave  = threadIdx.x >> 5;
    const int gwave = blockIdx.x * 8 + wave;      // [0, 512)
    const int gm0   = gwave << 4;                 // row block
    const int half  = lane >> 4;
    const int mrow  = lane & 15;
    const int ncol  = lane & 15;

    v16bf af[8];
    load_a_frags(Al + (size_t)(gm0 + mrow) * C_, half, af);

    double localsum = 0.0;
    for (int ns = 0; ns < M_; ns += 16) {
        const __bf16* brow = BlT + (size_t)(ns + ncol) * C_;
        v8f acc = wmma_k256(af, brow, half);

        const int n_g = ns + ncol;
        const int k   = n_g >> 3;
        #pragma unroll
        for (int v = 0; v < 8; ++v) {
            const int m_g = gm0 + v + 8 * half;
            float logit = (m_g == n_g) ? (NEGF * INV_T) : acc[v] * INV_T;
            float s = __expf(logit);
            s += __shfl_xor(s, 1, 32);
            s += __shfl_xor(s, 2, 32);
            s += __shfl_xor(s, 4, 32);            // group-of-8 sum of exps
            const float posl = P[(size_t)m_g * HW_ + k] * INV_T;
            const float tot  = s + __expf(posl);
            localsum += (double)(__logf(tot) - posl);   // counted 8x/group
        }
    }

    __shared__ double red[256];
    red[threadIdx.x] = localsum;
    __syncthreads();
    #pragma unroll
    for (int off = 128; off > 0; off >>= 1) {
        if (threadIdx.x < off) red[threadIdx.x] += red[threadIdx.x + off];
        __syncthreads();
    }
    if (threadIdx.x == 0) partials[blockIdx.x] = red[0];
}

// ---------------------------------------------------------------------------
// Kernel E: final deterministic reduction.  mean over N = B*HW*HW rows,
// and undo the 8x lane redundancy from kernel D.
// ---------------------------------------------------------------------------
__global__ void finalize_kernel(const double* __restrict__ partials,
                                float* __restrict__ out)
{
    __shared__ double red[64];
    const int t = threadIdx.x;
    red[t] = partials[t];
    __syncthreads();
    #pragma unroll
    for (int off = 32; off > 0; off >>= 1) {
        if (t < off) red[t] += red[t + off];
        __syncthreads();
    }
    if (t == 0) out[0] = (float)(red[0] * (1.0 / (8.0 * 8388608.0)));
}

// ---------------------------------------------------------------------------
extern "C" void kernel_launch(void* const* d_in, const int* in_sizes, int n_in,
                              void* d_out, int out_size, void* d_ws, size_t ws_size,
                              hipStream_t stream)
{
    const float* fq = (const float*)d_in[0];
    const float* fk = (const float*)d_in[1];
    float* out = (float*)d_out;

    char* ws = (char*)d_ws;
    const size_t OFF_INVQ = 0;                       //  32 KB
    const size_t OFF_INVK = 32768;                   //  32 KB
    const size_t OFF_AP   = 65536;                   //   4 MB
    const size_t OFF_BPT  = OFF_AP  + 4194304;       //   4 MB
    const size_t OFF_AL   = OFF_BPT + 4194304;       //   4 MB
    const size_t OFF_BLT  = OFF_AL  + 4194304;       //   4 MB
    const size_t OFF_P    = OFF_BLT + 4194304;       //  32 MB
    const size_t OFF_PART = OFF_P   + 33554432;      // 512 B
    if (ws_size < OFF_PART + 512) return;

    float*  invq = (float*)(ws + OFF_INVQ);
    float*  invk = (float*)(ws + OFF_INVK);
    __bf16* Ap   = (__bf16*)(ws + OFF_AP);
    __bf16* BpT  = (__bf16*)(ws + OFF_BPT);
    __bf16* Al   = (__bf16*)(ws + OFF_AL);
    __bf16* BlT  = (__bf16*)(ws + OFF_BLT);
    float*  P    = (float*)(ws + OFF_P);
    double* part = (double*)(ws + OFF_PART);

    norms_kernel   <<<  32, 256, 0, stream>>>(fq, fk, invq, invk);
    prep_kernel    <<<8192, 256, 0, stream>>>(fq, fk, invq, invk, Ap, BpT, Al, BlT);
    pos_gemm_kernel<<<  64, 256, 0, stream>>>(Ap, BpT, P);
    lneg_lse_kernel<<<  64, 256, 0, stream>>>(Al, BlT, P, part);
    finalize_kernel<<<   1,  64, 0, stream>>>(part, out);
}
